// MultiHeadAttention_60172491817303
// MI455X (gfx1250) — matrix operand order repaired and checked by reading, not compiled, not run on hardware
//
#include <hip/hip_runtime.h>
#include <hip/hip_bf16.h>
#include <stdint.h>

// ---------------- problem constants ----------------
#define IN_DIM  1024
#define DK      1024
#define NHEADS  16
#define HDIM    64
#define BATCH   2
#define SEQ     2048
#define MROWS   (BATCH * SEQ)   // 4096 tokens

// ---------------- register types (CDNA5 WMMA) ----------------
typedef __attribute__((ext_vector_type(16))) __bf16        v16bf;
typedef __attribute__((ext_vector_type(8)))  float         v8f;
typedef __attribute__((ext_vector_type(4)))  unsigned int  v4u;
typedef __attribute__((ext_vector_type(4)))  int           v4i;

#define AS1 __attribute__((address_space(1)))
#define AS3 __attribute__((address_space(3)))

#if defined(__gfx1250__) && __has_builtin(__builtin_amdgcn_global_load_async_to_lds_b128)
#define HAVE_ASYNC_LDS 1
#else
#define HAVE_ASYNC_LDS 0
#endif

union ABReg {
    v16bf        v;
    v4u          q[2];
    unsigned int u[8];
};

__device__ __forceinline__ unsigned int bf16_bits(float x) {
    unsigned int u = __float_as_uint(x);
    return (u + 0x7FFFu + ((u >> 16) & 1u)) >> 16;   // RNE
}
__device__ __forceinline__ unsigned int pack_bf16x2(float lo, float hi) {
    return (bf16_bits(lo) & 0xFFFFu) | (bf16_bits(hi) << 16);
}

// ---------------- conversion kernels ----------------
__global__ void cvt_f32_bf16(const float* __restrict__ src,
                             unsigned short* __restrict__ dst, int n) {
    int i = blockIdx.x * blockDim.x + threadIdx.x;
    if (i < n) dst[i] = (unsigned short)bf16_bits(src[i]);
}

// Wt[j][k] = bf16(W[k][j])  (1024x1024)
__global__ void transpose_w(const float* __restrict__ W,
                            unsigned short* __restrict__ Wt) {
    int idx = blockIdx.x * blockDim.x + threadIdx.x;
    int k = idx >> 10;
    int j = idx & 1023;
    Wt[j * 1024 + k] = (unsigned short)bf16_bits(W[k * 1024 + j]);
}

// ---------------- WMMA GEMM: Y = A[4096,1024] @ Bt^T + bias ----------------
// A: bf16 row-major (k contiguous).  Bt: bf16, Bt[j][k] (output-col major,
// k contiguous -> exactly the B-operand layout WMMA wants per lane).
// mode 0/1 : dst bf16 [b,h,s,d]   (Q / K)
// mode 2   : dst bf16 [b,h,d,s]   (V transposed per head)
// mode 3   : dst f32  [4096,1024] (final output)
__global__ __launch_bounds__(32)
void gemm_wmma(const unsigned short* __restrict__ A,
               const unsigned short* __restrict__ Bt,
               const float* __restrict__ bias,
               void* __restrict__ dst, int mode) {
    const int lane = threadIdx.x & 31;
    const int h2   = lane >> 4;        // wave half
    const int n16  = lane & 15;
    const int ntile = blockIdx.x;      // 0..63
    const int mtile = blockIdx.y;      // 0..255

    const unsigned short* aRow = A  + (size_t)(mtile * 16 + n16) * 1024;
    const unsigned short* bRow = Bt + (size_t)(ntile * 16 + n16) * 1024;

    v8f acc = {};
    for (int kk = 0; kk < 1024; kk += 32) {
        ABReg a, b;
        a.q[0] = *(const v4u*)(aRow + kk + h2 * 8);
        a.q[1] = *(const v4u*)(aRow + kk + 16 + h2 * 8);
        b.q[0] = *(const v4u*)(bRow + kk + h2 * 8);
        b.q[1] = *(const v4u*)(bRow + kk + 16 + h2 * 8);
        __builtin_prefetch(aRow + kk + 128, 0, 3);
        __builtin_prefetch(bRow + kk + 128, 0, 3);
        acc = __builtin_amdgcn_wmma_f32_16x16x32_bf16(
                  false, a.v, false, b.v, (short)0, acc, false, false);
    }

    const int   j    = ntile * 16 + n16;     // output feature
    const float bj   = bias[j];
    const int   head = j >> 6;
    const int   d    = j & 63;

#pragma unroll
    for (int v = 0; v < 8; ++v) {
        const int srow = mtile * 16 + h2 * 8 + v;   // token index
        const float val = acc[v] + bj;
        if (mode == 3) {
            ((float*)dst)[(size_t)srow * 1024 + j] = val;
        } else {
            const int b_ = srow >> 11;
            const int s  = srow & 2047;
            const unsigned short bits = (unsigned short)bf16_bits(val);
            if (mode == 2)   // V^T: [b,h,d,s]
                ((unsigned short*)dst)[((size_t)((b_ * NHEADS + head) * HDIM) + d) * SEQ + s] = bits;
            else             // Q/K: [b,h,s,d]
                ((unsigned short*)dst)[((size_t)((b_ * NHEADS + head) * SEQ) + s) * HDIM + d] = bits;
        }
    }
}

// ---------------- flash attention (per 16-q-row tile per wave) ----------------
// Block = 128 threads (4 waves). Block handles one (b,h) and 64 q rows.
// Per 32-key chunk: stage K[32,64] and Vt[64,32] into LDS with the CDNA5
// async global->LDS engine (ASYNCcnt), compute S^T = K@Q^T (two 16-t
// subtiles), online softmax (per-lane stats on column q), then the exp'd
// S^T tiles ARE the A operand of P@V with no cross-lane data movement.
__global__ __launch_bounds__(128)
void attention_wmma(const unsigned short* __restrict__ Q,
                    const unsigned short* __restrict__ K,
                    const unsigned short* __restrict__ Vt,
                    unsigned short* __restrict__ AO) {
    __shared__ unsigned short kTile[32 * HDIM];  // [t][d], d contiguous
    __shared__ unsigned short vTile[HDIM * 32];  // [d][t], t contiguous

    const int lane = threadIdx.x & 31;
    const int wave = threadIdx.x >> 5;
    const int h2   = lane >> 4;
    const int n16  = lane & 15;

    const int bh    = blockIdx.x >> 5;    // b*16 + h  (0..31)
    const int qg    = blockIdx.x & 31;    // 32 groups of 64 q rows
    const int qbase = qg * 64 + wave * 16;

    const unsigned short* Qh = Q  + (size_t)bh * SEQ * HDIM;
    const unsigned short* Kh = K  + (size_t)bh * SEQ * HDIM;
    const unsigned short* Vh = Vt + (size_t)bh * HDIM * SEQ;

    // Q tile as B operand: lane = column q, d in table order.  Two d-steps of 32.
    ABReg qB0, qB1;
    {
        const unsigned short* qp = Qh + (size_t)(qbase + n16) * HDIM;
        qB0.q[0] = *(const v4u*)(qp + h2 * 8);
        qB0.q[1] = *(const v4u*)(qp + 16 + h2 * 8);
        qB1.q[0] = *(const v4u*)(qp + 32 + h2 * 8);
        qB1.q[1] = *(const v4u*)(qp + 48 + h2 * 8);
    }

    v8f o[4] = {v8f{}, v8f{}, v8f{}, v8f{}};
    float mrow = -3.0e38f, lsum = 0.f;

    for (int t0 = 0; t0 < SEQ; t0 += 32) {
        __syncthreads();   // all waves done reading previous chunk
        {
            const int i0 = threadIdx.x, i1 = threadIdx.x + 128;
#if HAVE_ASYNC_LDS
            // K chunk: one contiguous 4KB block; V^T chunk: 64 rows x 64B.
            __builtin_amdgcn_global_load_async_to_lds_b128(
                (AS1 v4i*)(Kh + (size_t)t0 * HDIM + i0 * 8),
                (AS3 v4i*)(kTile + i0 * 8), 0, 0);
            __builtin_amdgcn_global_load_async_to_lds_b128(
                (AS1 v4i*)(Kh + (size_t)t0 * HDIM + i1 * 8),
                (AS3 v4i*)(kTile + i1 * 8), 0, 0);
            __builtin_amdgcn_global_load_async_to_lds_b128(
                (AS1 v4i*)(Vh + (size_t)(i0 >> 2) * SEQ + t0 + (i0 & 3) * 8),
                (AS3 v4i*)(vTile + i0 * 8), 0, 0);
            __builtin_amdgcn_global_load_async_to_lds_b128(
                (AS1 v4i*)(Vh + (size_t)(i1 >> 2) * SEQ + t0 + (i1 & 3) * 8),
                (AS3 v4i*)(vTile + i1 * 8), 0, 0);
#if __has_builtin(__builtin_amdgcn_s_wait_asynccnt)
            __builtin_amdgcn_s_wait_asynccnt(0);
#else
            asm volatile("s_wait_asynccnt 0x0" ::: "memory");
#endif
#else
            const v4u* srcK = (const v4u*)(Kh + (size_t)t0 * HDIM);
            v4u* dk = (v4u*)kTile;
            dk[i0] = srcK[i0];
            dk[i1] = srcK[i1];
            v4u* dv = (v4u*)vTile;
            dv[i0] = *((const v4u*)(Vh + (size_t)(i0 >> 2) * SEQ + t0) + (i0 & 3));
            dv[i1] = *((const v4u*)(Vh + (size_t)(i1 >> 2) * SEQ + t0) + (i1 & 3));
#endif
        }
        __syncthreads();   // chunk staged for all waves

        // ---- S^T subtiles: c0 -> keys [t0,t0+16), c1 -> [t0+16,t0+32) ----
        v8f c0 = {}, c1 = {};
        {
            ABReg a;
            const unsigned short* kr0 = kTile + n16 * HDIM;
            a.q[0] = *(const v4u*)(kr0 + h2 * 8);
            a.q[1] = *(const v4u*)(kr0 + 16 + h2 * 8);
            c0 = __builtin_amdgcn_wmma_f32_16x16x32_bf16(false, a.v, false, qB0.v, (short)0, c0, false, false);
            a.q[0] = *(const v4u*)(kr0 + 32 + h2 * 8);
            a.q[1] = *(const v4u*)(kr0 + 48 + h2 * 8);
            c0 = __builtin_amdgcn_wmma_f32_16x16x32_bf16(false, a.v, false, qB1.v, (short)0, c0, false, false);
            const unsigned short* kr1 = kTile + (16 + n16) * HDIM;
            a.q[0] = *(const v4u*)(kr1 + h2 * 8);
            a.q[1] = *(const v4u*)(kr1 + 16 + h2 * 8);
            c1 = __builtin_amdgcn_wmma_f32_16x16x32_bf16(false, a.v, false, qB0.v, (short)0, c1, false, false);
            a.q[0] = *(const v4u*)(kr1 + 32 + h2 * 8);
            a.q[1] = *(const v4u*)(kr1 + 48 + h2 * 8);
            c1 = __builtin_amdgcn_wmma_f32_16x16x32_bf16(false, a.v, false, qB1.v, (short)0, c1, false, false);
        }

        // ---- online softmax; lane holds column q = n16, rows t = h2*8+v ----
        float cmax = -3.0e38f;
        float p0[8], p1[8];
#pragma unroll
        for (int v = 0; v < 8; ++v) {
            c0[v] *= 0.125f;  // 1/sqrt(64)
            c1[v] *= 0.125f;
            cmax = fmaxf(cmax, fmaxf(c0[v], c1[v]));
        }
        cmax = fmaxf(cmax, __shfl_xor(cmax, 16, 32));   // combine lane halves
        const float mnew  = fmaxf(mrow, cmax);
        const float scale = __expf(mrow - mnew);
        float csum = 0.f;
#pragma unroll
        for (int v = 0; v < 8; ++v) {
            p0[v] = __expf(c0[v] - mnew);
            p1[v] = __expf(c1[v] - mnew);
            csum += p0[v] + p1[v];
        }
        csum += __shfl_xor(csum, 16, 32);
        lsum = lsum * scale + csum;
        mrow = mnew;

        // ---- exp'd S^T tiles are exactly the A operand of P[16q x 32t] ----
        ABReg pa;
#pragma unroll
        for (int v = 0; v < 4; ++v) {
            pa.u[v]     = pack_bf16x2(p0[2 * v], p0[2 * v + 1]);
            pa.u[v + 4] = pack_bf16x2(p1[2 * v], p1[2 * v + 1]);
        }

        // ---- rescale O accumulators (row q = h2*8+v; stats live in lane q) ----
#pragma unroll
        for (int v = 0; v < 8; ++v) {
            const float sv = __shfl(scale, h2 * 8 + v, 32);
            o[0][v] *= sv; o[1][v] *= sv; o[2][v] *= sv; o[3][v] *= sv;
        }

        // ---- O += P @ V : B operand from vTile (col d, contiguous t) ----
#pragma unroll
        for (int dt = 0; dt < 4; ++dt) {
            ABReg b;
            const unsigned short* vr = vTile + (dt * 16 + n16) * 32 + h2 * 8;
            b.q[0] = *(const v4u*)(vr);
            b.q[1] = *(const v4u*)(vr + 16);
            o[dt] = __builtin_amdgcn_wmma_f32_16x16x32_bf16(
                        false, pa.v, false, b.v, (short)0, o[dt], false, false);
        }
    }

    // ---- normalize and store to AO[b, s, h*64+d] (A-layout for out-proj) ----
    const float linv = 1.0f / lsum;
    const int b_   = bh >> 4;
    const int head = bh & 15;
#pragma unroll
    for (int v = 0; v < 8; ++v) {
        const float sv = __shfl(linv, h2 * 8 + v, 32);
        const int qrow  = qbase + h2 * 8 + v;
        unsigned short* dst = AO + (size_t)(b_ * SEQ + qrow) * 1024 + head * HDIM;
#pragma unroll
        for (int dt = 0; dt < 4; ++dt)
            dst[dt * 16 + n16] = (unsigned short)bf16_bits(o[dt][v] * sv);
    }
}

// ---------------- host launch ----------------
extern "C" void kernel_launch(void* const* d_in, const int* in_sizes, int n_in,
                              void* d_out, int out_size, void* d_ws, size_t ws_size,
                              hipStream_t stream) {
    (void)in_sizes; (void)n_in; (void)out_size; (void)ws_size;
    const float* x  = (const float*)d_in[0];
    const float* Wq = (const float*)d_in[1];
    const float* bq = (const float*)d_in[2];
    const float* Wk = (const float*)d_in[3];
    const float* bk = (const float*)d_in[4];
    const float* Wv = (const float*)d_in[5];
    const float* bv = (const float*)d_in[6];
    const float* Wo = (const float*)d_in[7];
    const float* bo = (const float*)d_in[8];
    float* out = (float*)d_out;

    char* ws = (char*)d_ws;
    size_t off = 0;
    auto alloc = [&](size_t bytes) -> char* {
        char* p = ws + off;
        off += (bytes + 255) & ~(size_t)255;
        return p;
    };
    unsigned short* xb  = (unsigned short*)alloc((size_t)MROWS * 1024 * 2);
    unsigned short* WtQ = (unsigned short*)alloc((size_t)1024 * 1024 * 2);
    unsigned short* WtK = (unsigned short*)alloc((size_t)1024 * 1024 * 2);
    unsigned short* WtV = (unsigned short*)alloc((size_t)1024 * 1024 * 2);
    unsigned short* WtO = (unsigned short*)alloc((size_t)1024 * 1024 * 2);
    unsigned short* Qb  = (unsigned short*)alloc((size_t)MROWS * 1024 * 2);
    unsigned short* Kb  = (unsigned short*)alloc((size_t)MROWS * 1024 * 2);
    unsigned short* Vtb = (unsigned short*)alloc((size_t)MROWS * 1024 * 2);
    unsigned short* AOb = (unsigned short*)alloc((size_t)MROWS * 1024 * 2);

    const int n_x = MROWS * 1024;
    cvt_f32_bf16<<<n_x / 256, 256, 0, stream>>>(x, xb, n_x);
    transpose_w<<<(1024 * 1024) / 256, 256, 0, stream>>>(Wq, WtQ);
    transpose_w<<<(1024 * 1024) / 256, 256, 0, stream>>>(Wk, WtK);
    transpose_w<<<(1024 * 1024) / 256, 256, 0, stream>>>(Wv, WtV);
    transpose_w<<<(1024 * 1024) / 256, 256, 0, stream>>>(Wo, WtO);

    dim3 ggrid(64, 256);  // N/16 x M/16
    gemm_wmma<<<ggrid, 32, 0, stream>>>(xb, WtQ, bq, Qb, 0);
    gemm_wmma<<<ggrid, 32, 0, stream>>>(xb, WtK, bk, Kb, 1);
    gemm_wmma<<<ggrid, 32, 0, stream>>>(xb, WtV, bv, Vtb, 2);

    attention_wmma<<<1024, 128, 0, stream>>>(Qb, Kb, Vtb, AOb);

    gemm_wmma<<<ggrid, 32, 0, stream>>>(AOb, WtO, bo, out, 3);
}
